// SDFNetwork_2d_hash_fix_61203283968105
// MI455X (gfx1250) — matrix operand order repaired and checked by reading, not compiled, not run on hardware
//
#include <hip/hip_runtime.h>
#include <hip/hip_bf16.h>

typedef __attribute__((ext_vector_type(16))) _Float16 v16h;
typedef __attribute__((ext_vector_type(8)))  float    v8f;

#define N_PTS   131072
#define N_EGO   2048
#define NCLUS   4
#define TSIZE   524288u            // 1<<19
#define HPRIME  2654435761u
#define PI_F    3.14159265358979323846f

// ---- workspace layout (bytes) ----
// [0,16)                : 4 int cluster counters
// [256, 256+65536)      : f16 weight blob: 4 clusters x 16 tiles x 32 lanes x 16 halfs
// [65792, 65792+4*N*4)  : perm lists, one N-sized region per cluster
#define WS_BLOB_OFF  256
#define WS_PERM_OFF  65792
#define BLOB_CLUSTER_BYTES (16 * 32 * 16 * 2)   // 16 KB per cluster

// A-tile ids per cluster: 0..3 = W1 rows 16t (K=12 pad 32)
//                         4..11 = W2 (out-tile=(t-4)>>1, kchunk=(t-4)&1)
//                         12..15 = tiny_w rows 1+16t (K=32)

__global__ void zero_counts_kernel(int* counts) {
    if (threadIdx.x < NCLUS) counts[threadIdx.x] = 0;
}

// One thread per (cluster, tile, lane); writes 16 f16 values in this file's contiguous-k
// per-lane A-matrix layout: a[j] holds W[M, k], k = 16*hi + j.
__global__ __launch_bounds__(256) void pack_weights_kernel(
    const float* __restrict__ p_w1,    // (C,64,12)
    const float* __restrict__ p_w2,    // (C,64,64)
    const float* __restrict__ tiny_w,  // (C,65,32)
    _Float16* __restrict__ blob)
{
    int gid = blockIdx.x * 256 + threadIdx.x;          // 0..2047
    int c    = gid >> 9;
    int rem  = gid & 511;
    int tile = rem >> 5;
    int lane = rem & 31;
    int hi   = lane >> 4;
    int M    = lane & 15;
    #pragma unroll
    for (int j = 0; j < 16; ++j) {
        int k = 16 * hi + j;
        float v = 0.0f;
        if (tile < 4) {
            int Mg = 16 * tile + M;
            if (k < 12) v = p_w1[(c * 64 + Mg) * 12 + k];
        } else if (tile < 12) {
            int ot = (tile - 4) >> 1;
            int kc = (tile - 4) & 1;
            int Mg = 16 * ot + M;
            v = p_w2[(c * 64 + Mg) * 64 + kc * 32 + k];
        } else {
            int Mg = 1 + 16 * (tile - 12) + M;         // skip row 0 (x[:,1:])
            v = tiny_w[(c * 65 + Mg) * 32 + k];
        }
        blob[(size_t)gid * 16 + j] = (_Float16)v;
    }
}

// NN search + partition: ego points (+0.5*|e|^2) staged in LDS; uniform j loop
// -> LDS broadcast reads. argmin of 0.5|e|^2 - w.e  (same argmin as reference).
__global__ __launch_bounds__(256) void nn_partition_kernel(
    const float* __restrict__ inputs,
    const float* __restrict__ ego,
    const int*   __restrict__ ego2cluster,
    const float* __restrict__ S,          // scale_mat row-major 3x3
    int* __restrict__ counts,
    int* __restrict__ perm)
{
    __shared__ float4 se[N_EGO];
    int tid = threadIdx.x;
    for (int j = tid; j < N_EGO; j += 256) {
        float ex = ego[j * 3 + 0], ey = ego[j * 3 + 1], ez = ego[j * 3 + 2];
        se[j] = make_float4(ex, ey, ez, 0.5f * (ex * ex + ey * ey + ez * ez));
    }
    __syncthreads();

    int p = blockIdx.x * 256 + tid;                    // N multiple of 256
    float x0 = inputs[p * 3 + 0], x1 = inputs[p * 3 + 1], x2 = inputs[p * 3 + 2];
    float w0 = x0 * S[0] + x1 * S[3] + x2 * S[6];
    float w1 = x0 * S[1] + x1 * S[4] + x2 * S[7];
    float w2 = x0 * S[2] + x1 * S[5] + x2 * S[8];

    float best = 3.4e38f; int bi = 0;
    #pragma unroll 4
    for (int j = 0; j < N_EGO; ++j) {
        float4 e = se[j];
        float sc = e.w - (w0 * e.x + w1 * e.y + w2 * e.z);
        if (sc < best) { best = sc; bi = j; }          // first-min like argmin
    }
    int c    = ego2cluster[bi];
    int slot = atomicAdd(&counts[c], 1);
    perm[c * N_PTS + slot] = p;
}

// Fast sigmoid: v_exp_f32 + v_rcp_f32 (avoids the IEEE v_div_scale sequence).
__device__ __forceinline__ float sigf(float x) {
    return __builtin_amdgcn_rcpf(1.0f + __expf(-x));
}

union B16u { v16h v; unsigned u[8]; };
union H8u  { _Float16 h[8]; unsigned u[4]; };
union AVu  { int4 q[2]; v16h v; };

// One wave = 16 points of one cluster. All three GEMMs as W @ X^T via
// v_wmma_f32_16x16x32_f16; D(feature x point) chains into next B with an
// f16 pack + shfl_xor(16) half-wave exchange, no LDS round trip.
// Per-block: the cluster's 16KB A-tile blob is staged global->LDS with
// async b128 copies (ASYNCcnt) since all 8 waves share one cluster.
__global__ __launch_bounds__(256) void mlp_kernel(
    const float* __restrict__ inputs,
    const int*   __restrict__ counts,
    const int*   __restrict__ perm,
    const _Float16* __restrict__ blob,
    const float* __restrict__ hash_tables, // (C,16,TSIZE,2)
    const float* __restrict__ tiny_b,      // (C,65)
    const float* __restrict__ b1,          // (C,64)
    const float* __restrict__ b2,          // (C,64)
    const float* __restrict__ w3,          // (C,1,64)
    const float* __restrict__ b3,          // (C,1)
    float* __restrict__ out)               // (N,65)
{
    __shared__ alignas(32) char swb[BLOB_CLUSTER_BYTES];   // 16 KB A-tiles

    int c    = blockIdx.y;
    int lane = threadIdx.x & 31;
    int wave = threadIdx.x >> 5;
    int cnt  = counts[c];
    int base = (blockIdx.x * 8 + wave) * 16;

    // ---- stage this cluster's weight blob into LDS (async DMA path) ----
    {
        unsigned long long gbase =
            (unsigned long long)(uintptr_t)((const char*)blob + (size_t)c * BLOB_CLUSTER_BYTES);
        unsigned goff = threadIdx.x * 16u;                 // 256 thr x 4 x 16B = 16KB
        #pragma unroll
        for (int it = 0; it < BLOB_CLUSTER_BYTES / (256 * 16); ++it) {
            unsigned ldsaddr = (unsigned)(uintptr_t)(&swb[goff]);
            asm volatile("global_load_async_to_lds_b128 %0, %1, %2"
                         :: "v"(ldsaddr), "v"(goff), "s"(gbase)
                         : "memory");
            goff += 256u * 16u;
        }
        asm volatile("s_wait_asynccnt 0" ::: "memory");
    }
    __syncthreads();

    if (base >= cnt) return;                           // wave-uniform exit

    int hi = lane >> 4;
    int pl = lane & 15;
    int j  = base + pl; if (j >= cnt) j = cnt - 1;     // tail clamp (benign dup)
    int p  = perm[c * N_PTS + j];

    float x0 = inputs[p * 3 + 0], x1 = inputs[p * 3 + 1], x2 = inputs[p * 3 + 2];

    // ---------------- layer 1: freq-encode -> B (K=32, feats 0..11, rest 0)
    // B layout: lanes 0-15 hold K=0..15 (all 12 feats of their point),
    // lanes 16-31 hold K=16..31 (all zero padding).
    v16h bf;
    #pragma unroll
    for (int i = 0; i < 16; ++i) bf[i] = (_Float16)0.0f;
    if (hi == 0) {
        float xy[2] = { x0, x1 };
        #pragma unroll
        for (int d = 0; d < 2; ++d) {
            #pragma unroll
            for (int r = 0; r < 3; ++r) {
                float ang = xy[d] * (PI_F * (float)(1 << r));
                bf[d * 6 + r]     = (_Float16)__sinf(ang);
                bf[d * 6 + 3 + r] = (_Float16)__cosf(ang);
            }
        }
    }

    H8u pk1[4];                                        // sigmoid(L1) packed f16
    #pragma unroll
    for (int t = 0; t < 4; ++t) {
        AVu a;
        a.q[0] = *(const int4*)(swb + (t * 32 + lane) * 32);
        a.q[1] = *(const int4*)(swb + (t * 32 + lane) * 32 + 16);
        v8f cc;
        #pragma unroll
        for (int i = 0; i < 8; ++i) cc[i] = b1[c * 64 + 16 * t + 8 * hi + i];
        cc = __builtin_amdgcn_wmma_f32_16x16x32_f16(false, a.v, false, bf,
                                                    (short)0, cc, false, false);
        #pragma unroll
        for (int i = 0; i < 8; ++i) pk1[t].h[i] = (_Float16)sigf(cc[i]);
    }

    // Build layer-2 B chunks (K=64 -> two 32-K chunks) from D tiles:
    // lo lane needs rows 0..15 of tiles {0,2}; hi lane rows 0..15 of {1,3}.
    B16u bA, bB;
    #pragma unroll
    for (int k = 0; k < 4; ++k) {
        unsigned o0 = (unsigned)__shfl_xor((int)pk1[0].u[k], 16);
        unsigned o1 = (unsigned)__shfl_xor((int)pk1[1].u[k], 16);
        unsigned o2 = (unsigned)__shfl_xor((int)pk1[2].u[k], 16);
        unsigned o3 = (unsigned)__shfl_xor((int)pk1[3].u[k], 16);
        bA.u[k]     = hi ? o1           : pk1[0].u[k]; // K 0..7 | 16..23
        bA.u[4 + k] = hi ? pk1[1].u[k]  : o0;          // K 8..15| 24..31
        bB.u[k]     = hi ? o3           : pk1[2].u[k];
        bB.u[4 + k] = hi ? pk1[3].u[k]  : o2;
    }

    // ---------------- layer 2: 64->64, two WMMAs per 16-out tile
    float s2[4][8];
    #pragma unroll
    for (int ot = 0; ot < 4; ++ot) {
        AVu a0, a1;
        a0.q[0] = *(const int4*)(swb + ((4 + 2 * ot) * 32 + lane) * 32);
        a0.q[1] = *(const int4*)(swb + ((4 + 2 * ot) * 32 + lane) * 32 + 16);
        a1.q[0] = *(const int4*)(swb + ((5 + 2 * ot) * 32 + lane) * 32);
        a1.q[1] = *(const int4*)(swb + ((5 + 2 * ot) * 32 + lane) * 32 + 16);
        v8f cc;
        #pragma unroll
        for (int i = 0; i < 8; ++i) cc[i] = b2[c * 64 + 16 * ot + 8 * hi + i];
        cc = __builtin_amdgcn_wmma_f32_16x16x32_f16(false, a0.v, false, bA.v,
                                                    (short)0, cc, false, false);
        cc = __builtin_amdgcn_wmma_f32_16x16x32_f16(false, a1.v, false, bB.v,
                                                    (short)0, cc, false, false);
        #pragma unroll
        for (int i = 0; i < 8; ++i) s2[ot][i] = sigf(cc[i]);
    }

    // ---------------- layer 3: prior_z = w3 . h2 + b3 (half-wave dot)
    float part = 0.0f;
    #pragma unroll
    for (int ot = 0; ot < 4; ++ot)
        #pragma unroll
        for (int i = 0; i < 8; ++i)
            part += s2[ot][i] * w3[c * 64 + 16 * ot + 8 * hi + i];
    float prior_z = part + __shfl_xor(part, 16) + b3[c];

    // ---------------- hash encode -> B (K=32 = 16 levels x 2 feats)
    // lo lane: levels 0..7 (K 0..15); hi lane: levels 8..15 (K 16..31).
    float u0 = x0 * (1.0f / 30.0f) + 0.5f;
    float u1 = x1 * (1.0f / 30.0f) + 0.5f;
    v16h bh;
    float sc = hi ? (16.0f * 25.62890625f) : 16.0f;    // 16 * 1.5^(8*hi)
    #pragma unroll
    for (int q = 0; q < 8; ++q) {
        int lvl = 8 * hi + q;
        float px = u0 * sc, py = u1 * sc;
        float fx = floorf(px), fy = floorf(py);
        float wx = px - fx,   wy = py - fy;
        unsigned cx = (unsigned)fx, cy = (unsigned)fy;
        const float2* tab = (const float2*)hash_tables + (size_t)(c * 16 + lvl) * TSIZE;
        unsigned hy0 = cy * HPRIME, hy1 = (cy + 1u) * HPRIME;
        float2 v00 = tab[( cx        ^ hy0) & (TSIZE - 1u)];
        float2 v10 = tab[((cx + 1u)  ^ hy0) & (TSIZE - 1u)];
        float2 v01 = tab[( cx        ^ hy1) & (TSIZE - 1u)];
        float2 v11 = tab[((cx + 1u)  ^ hy1) & (TSIZE - 1u)];
        float w00 = (1.0f - wx) * (1.0f - wy), w10 = wx * (1.0f - wy);
        float w01 = (1.0f - wx) * wy,          w11 = wx * wy;
        bh[2 * q]     = (_Float16)(v00.x * w00 + v10.x * w10 + v01.x * w01 + v11.x * w11);
        bh[2 * q + 1] = (_Float16)(v00.y * w00 + v10.y * w10 + v01.y * w01 + v11.y * w11);
        sc *= 1.5f;
    }

    // ---------------- tiny layer: rows 1..64 of tiny_w, store output
    size_t orow = (size_t)p * 65;
    #pragma unroll
    for (int t = 0; t < 4; ++t) {
        AVu a;
        a.q[0] = *(const int4*)(swb + ((12 + t) * 32 + lane) * 32);
        a.q[1] = *(const int4*)(swb + ((12 + t) * 32 + lane) * 32 + 16);
        v8f cc;
        #pragma unroll
        for (int i = 0; i < 8; ++i) cc[i] = 0.0f;
        cc = __builtin_amdgcn_wmma_f32_16x16x32_f16(false, a.v, false, bh,
                                                    (short)0, cc, false, false);
        int col0 = 1 + 16 * t + 8 * hi;
        #pragma unroll
        for (int i = 0; i < 8; ++i)
            out[orow + col0 + i] = cc[i] + tiny_b[c * 65 + col0 + i];
    }
    if (hi == 0) out[orow] = x2 - prior_z;             // z - prior_z (SCALE=1)
}

extern "C" void kernel_launch(void* const* d_in, const int* in_sizes, int n_in,
                              void* d_out, int out_size, void* d_ws, size_t ws_size,
                              hipStream_t stream) {
    const float* inputs      = (const float*)d_in[0];
    const float* ego_points  = (const float*)d_in[1];
    const int*   ego2cluster = (const int*)  d_in[2];
    const float* scale_mat   = (const float*)d_in[3];
    const float* hash_tables = (const float*)d_in[4];
    const float* tiny_w      = (const float*)d_in[5];
    const float* tiny_b      = (const float*)d_in[6];
    const float* p_w1        = (const float*)d_in[7];
    const float* p_b1        = (const float*)d_in[8];
    const float* p_w2        = (const float*)d_in[9];
    const float* p_b2        = (const float*)d_in[10];
    const float* p_w3        = (const float*)d_in[11];
    const float* p_b3        = (const float*)d_in[12];
    float* out = (float*)d_out;

    int*       counts = (int*)d_ws;
    _Float16*  blob   = (_Float16*)((char*)d_ws + WS_BLOB_OFF);
    int*       perm   = (int*)((char*)d_ws + WS_PERM_OFF);

    zero_counts_kernel<<<1, 32, 0, stream>>>(counts);
    pack_weights_kernel<<<8, 256, 0, stream>>>(p_w1, p_w2, tiny_w, blob);
    nn_partition_kernel<<<N_PTS / 256, 256, 0, stream>>>(
        inputs, ego_points, ego2cluster, scale_mat, counts, perm);
    dim3 grid(N_PTS / (16 * 8), NCLUS);
    mlp_kernel<<<grid, 256, 0, stream>>>(
        inputs, counts, perm, blob, hash_tables, tiny_b,
        p_b1, p_b2, p_w3, p_b3, out);
}
